// SupplyChainGNNWithAttention_81853486727228
// MI455X (gfx1250) — matrix operand order repaired and checked by reading, not compiled, not run on hardware
//
#include <hip/hip_runtime.h>
#include <hip/hip_bf16.h>
#include <math.h>

typedef __attribute__((ext_vector_type(16))) __bf16 v16bf;
typedef __attribute__((ext_vector_type(8)))  __bf16 v8bf;
typedef __attribute__((ext_vector_type(8)))  float  v8f;

#define LRELU_SLOPE 0.2f

__device__ __forceinline__ void atomicMaxF(float* addr, float v) {
    // monotone-bit-pattern trick; init value must be -FLT_MAX
    if (v >= 0.0f)
        atomicMax((int*)addr, __float_as_int(v));
    else
        atomicMin((unsigned int*)addr, __float_as_uint(v));
}

__device__ __forceinline__ v16bf cat8(v8bf lo, v8bf hi) {
    return __builtin_shufflevector(lo, hi, 0, 1, 2, 3, 4, 5, 6, 7,
                                           8, 9, 10, 11, 12, 13, 14, 15);
}

// ---------------------------------------------------------------------------
// GEMM: C[M,Nc] = A[M,K] * W[K,Nc], A pre-converted to bf16 row-major,
// W pre-transposed to bf16 Wt[Nc][K].  No LDS: fragments come straight from
// global as 128-bit loads per the ISA wave32 16-bit A/B layouts.
// grid = (ceil(M/64), Nc/64), block = 128 (4 waves); wave w owns rows
// m0 = (bx*4+w)*16 and all 64 columns of the block's N tile (4 accumulators).
// ---------------------------------------------------------------------------
template <int K>
__global__ __launch_bounds__(128)
void gemm_bf16_wmma(const __bf16* __restrict__ A, const __bf16* __restrict__ Wt,
                    float* __restrict__ C, int M, int Nc) {
    const int wave = threadIdx.x >> 5;
    const int lane = threadIdx.x & 31;
    const int m0   = (blockIdx.x * 4 + wave) << 4;
    if (m0 >= M) return;                      // wave-uniform: EXEC stays all-1s
    const int n0  = blockIdx.y << 6;
    const int idx = lane & 15;                // A row / B,C column within tile
    const int kh  = lane >> 4;                // K-half selector (ISA layout)

    const __bf16* arow = A + (size_t)(m0 + idx) * K;

    v8f acc[4] = {{}, {}, {}, {}};

    #pragma unroll
    for (int k0 = 0; k0 < K; k0 += 32) {
        // A frag: lane rows of K = kh*8+[0..7] and 16+kh*8+[0..7] (2x b128)
        v8bf alo = *(const v8bf*)(arow + k0 + kh * 8);
        v8bf ahi = *(const v8bf*)(arow + k0 + 16 + kh * 8);
        v16bf af = cat8(alo, ahi);
        #pragma unroll
        for (int j = 0; j < 4; ++j) {
            // B frag: column n0+j*16+idx, K = kh*8+[0..7] and 16+kh*8+[0..7] in Wt
            const __bf16* brow = Wt + (size_t)(n0 + j * 16 + idx) * K + k0 + kh * 8;
            v16bf bf = cat8(*(const v8bf*)brow, *(const v8bf*)(brow + 16));
            acc[j] = __builtin_amdgcn_wmma_f32_16x16x32_bf16(
                         false, af, false, bf, (short)0, acc[j], false, false);
        }
    }

    // D layout: element r of lane L -> M = r + 8*(L>>4), N = L&15
    #pragma unroll
    for (int j = 0; j < 4; ++j) {
        const int ccol = n0 + j * 16 + idx;
        #pragma unroll
        for (int r = 0; r < 8; ++r)
            C[(size_t)(m0 + r + (kh << 3)) * Nc + ccol] = acc[j][r];
    }
}

// ---------------------------------------------------------------------------
__global__ void f32_to_bf16(const float* __restrict__ in, __bf16* __restrict__ out,
                            int n2) {             // n2 = n/2, n even
    int i = blockIdx.x * blockDim.x + threadIdx.x;
    if (i >= n2) return;
    out[2 * i]     = (__bf16)in[2 * i];
    out[2 * i + 1] = (__bf16)in[2 * i + 1];
}

__global__ void w_transpose_bf16(const float* __restrict__ W, __bf16* __restrict__ Wt,
                                 int K, int Nc) {
    int i = blockIdx.x * blockDim.x + threadIdx.x;
    if (i >= K * Nc) return;
    int r = i / Nc, c = i % Nc;
    Wt[(size_t)c * K + r] = (__bf16)W[i];
}

__global__ void fill_f32(float* __restrict__ p, float v, int n) {
    int i = blockIdx.x * blockDim.x + threadIdx.x;
    if (i < n) p[i] = v;
}

// alpha_src[n,h] = <h[n,h,:], a_s[h,:]> ; same for alpha_dst
__global__ void alpha_kernel(const float* __restrict__ h,
                             const float* __restrict__ a_s,
                             const float* __restrict__ a_d,
                             float* __restrict__ als, float* __restrict__ ald,
                             int NH, int H, int C) {
    int idx = blockIdx.x * blockDim.x + threadIdx.x;
    if (idx >= NH) return;
    int n = idx / H, hh = idx % H;
    const float* rowp = h + (size_t)n * H * C + (size_t)hh * C;
    const float* asr = a_s + hh * C;
    const float* adr = a_d + hh * C;
    float s1 = 0.f, s2 = 0.f;
    for (int c = 0; c < C; ++c) {
        float v = rowp[c];
        s1 += v * asr[c];
        s2 += v * adr[c];
    }
    als[idx] = s1;
    ald[idx] = s2;
}

__device__ __forceinline__ void edge_sd(int i, const int* __restrict__ src,
                                        const int* __restrict__ dst, int E,
                                        int& s, int& d) {
    if (i < E) { s = src[i]; d = dst[i]; }
    else       { s = d = i - E; }          // self-loops appended after edges
}

__global__ void edge_max(const float* __restrict__ als, const float* __restrict__ ald,
                         const int* __restrict__ src, const int* __restrict__ dst,
                         float* __restrict__ m, int E, int N, int H) {
    int g = blockIdx.x * blockDim.x + threadIdx.x;
    if (g >= (E + N) * H) return;
    int i = g / H, hh = g % H, s, d;
    edge_sd(i, src, dst, E, s, d);
    float v = als[s * H + hh] + ald[d * H + hh];
    float e = v > 0.f ? v : LRELU_SLOPE * v;
    atomicMaxF(&m[d * H + hh], e);
}

__global__ void edge_den(const float* __restrict__ als, const float* __restrict__ ald,
                         const int* __restrict__ src, const int* __restrict__ dst,
                         const float* __restrict__ m, float* __restrict__ den,
                         int E, int N, int H) {
    int g = blockIdx.x * blockDim.x + threadIdx.x;
    if (g >= (E + N) * H) return;
    int i = g / H, hh = g % H, s, d;
    edge_sd(i, src, dst, E, s, d);
    float v = als[s * H + hh] + ald[d * H + hh];
    float e = v > 0.f ? v : LRELU_SLOPE * v;
    atomicAdd(&den[d * H + hh], expf(e - m[d * H + hh]));
}

// one block per (edge|self-loop), blockDim = H*C
__global__ void edge_scatter(const float* __restrict__ h,
                             const float* __restrict__ als, const float* __restrict__ ald,
                             const float* __restrict__ m,   const float* __restrict__ den,
                             const int* __restrict__ src,   const int* __restrict__ dst,
                             float* __restrict__ acc, int E, int N, int H, int C) {
    __shared__ float sattn[4];
    int i = blockIdx.x, t = threadIdx.x, s, d;
    edge_sd(i, src, dst, E, s, d);
    if (t < H) {
        float v = als[s * H + t] + ald[d * H + t];
        float e = v > 0.f ? v : LRELU_SLOPE * v;
        sattn[t] = expf(e - m[d * H + t]) / (den[d * H + t] + 1e-16f);
    }
    __syncthreads();
    int head = t / C;
    atomicAdd(&acc[(size_t)d * (H * C) + t],
              h[(size_t)s * (H * C) + t] * sattn[head]);
}

// out = act(acc + b); act: 0 = ELU, 1 = ReLU (in place)
__global__ void bias_act(float* __restrict__ p, const float* __restrict__ b,
                         int total, int HC, int act) {
    int i = blockIdx.x * blockDim.x + threadIdx.x;
    if (i >= total) return;
    float v = p[i] + b[i % HC];
    if (act == 0) v = v > 0.f ? v : (expf(v) - 1.0f);
    else          v = fmaxf(v, 0.f);
    p[i] = v;
}

// head: relu(h @ cW1 + cb1) @ cW2 + cb2 -> log_softmax, per node
__global__ void classifier(const float* __restrict__ h,
                           const float* __restrict__ cW1, const float* __restrict__ cb1,
                           const float* __restrict__ cW2, const float* __restrict__ cb2,
                           float* __restrict__ out, int N) {
    int n = blockIdx.x * blockDim.x + threadIdx.x;
    if (n >= N) return;
    const float* rowp = h + (size_t)n * 64;
    float t[32];
    #pragma unroll 4
    for (int j = 0; j < 32; ++j) {
        float s = cb1[j];
        for (int c = 0; c < 64; ++c) s += rowp[c] * cW1[c * 32 + j];
        t[j] = fmaxf(s, 0.f);
    }
    float l0 = cb2[0], l1 = cb2[1];
    for (int j = 0; j < 32; ++j) { l0 += t[j] * cW2[j * 2]; l1 += t[j] * cW2[j * 2 + 1]; }
    float mx  = fmaxf(l0, l1);
    float lse = mx + logf(expf(l0 - mx) + expf(l1 - mx));
    out[n * 2 + 0] = l0 - lse;
    out[n * 2 + 1] = l1 - lse;
}

// ---------------------------------------------------------------------------
static void run_edge_phase(const float* hbuf, float* accbuf, const float* a_s,
                           const float* a_d, const float* b, int H, int C, int act,
                           float* al_s, float* al_d, float* mbuf, float* dbuf,
                           const int* src, const int* dst, int N, int E,
                           hipStream_t stream) {
    const int HC = H * C;
    const int nh = N * H;
    alpha_kernel<<<(nh + 127) / 128, 128, 0, stream>>>(hbuf, a_s, a_d, al_s, al_d, nh, H, C);
    fill_f32<<<(nh + 255) / 256, 256, 0, stream>>>(mbuf, -3.402823466e38f, nh);
    fill_f32<<<(nh + 255) / 256, 256, 0, stream>>>(dbuf, 0.f, nh);

    const int nhc = N * HC;
    fill_f32<<<(nhc + 255) / 256, 256, 0, stream>>>(accbuf, 0.f, nhc);

    const int tot = (E + N) * H;
    edge_max<<<(tot + 255) / 256, 256, 0, stream>>>(al_s, al_d, src, dst, mbuf, E, N, H);
    edge_den<<<(tot + 255) / 256, 256, 0, stream>>>(al_s, al_d, src, dst, mbuf, dbuf, E, N, H);
    edge_scatter<<<E + N, HC, 0, stream>>>(hbuf, al_s, al_d, mbuf, dbuf, src, dst,
                                           accbuf, E, N, H, C);
    bias_act<<<(nhc + 255) / 256, 256, 0, stream>>>(accbuf, b, nhc, HC, act);
}

extern "C" void kernel_launch(void* const* d_in, const int* in_sizes, int n_in,
                              void* d_out, int out_size, void* d_ws, size_t ws_size,
                              hipStream_t stream) {
    (void)n_in; (void)out_size; (void)ws_size;
    const float* x   = (const float*)d_in[0];
    const int*  eidx = (const int*)d_in[1];
    const float* W1  = (const float*)d_in[2];
    const float* a1s = (const float*)d_in[3];
    const float* a1d = (const float*)d_in[4];
    const float* b1  = (const float*)d_in[5];
    const float* W2  = (const float*)d_in[6];
    const float* a2s = (const float*)d_in[7];
    const float* a2d = (const float*)d_in[8];
    const float* b2  = (const float*)d_in[9];
    const float* W3  = (const float*)d_in[10];
    const float* a3s = (const float*)d_in[11];
    const float* a3d = (const float*)d_in[12];
    const float* b3  = (const float*)d_in[13];
    const float* cW1 = (const float*)d_in[14];
    const float* cb1 = (const float*)d_in[15];
    const float* cW2 = (const float*)d_in[16];
    const float* cb2 = (const float*)d_in[17];
    float* out = (float*)d_out;

    const int N = in_sizes[0] / 128;   // 50000
    const int E = in_sizes[1] / 2;     // 500000
    const int* src = eidx;
    const int* dst = eidx + E;

    // workspace carving
    float*  bufA = (float*)d_ws;                   // N*256 f32 (GEMM output h)
    float*  bufB = bufA + (size_t)N * 256;         // N*256 f32 (aggregated / input)
    float*  al_s = bufB + (size_t)N * 256;         // N*4
    float*  al_d = al_s + (size_t)N * 4;           // N*4
    float*  mbuf = al_d + (size_t)N * 4;           // N*4
    float*  dbuf = mbuf + (size_t)N * 4;           // N*4
    __bf16* hbf  = (__bf16*)(dbuf + (size_t)N * 4);// N*256 bf16 (GEMM input)
    __bf16* wtb  = hbf + (size_t)N * 256;          // 256*256 bf16 (W transposed)

    const dim3 gblk(128);
    const int  mblocks = (N / 16 + 3) / 4;         // 4 wave-tiles of 16 rows per block

    // ---- Layer 1: x[N,128] @ W1[128,256] ----
    f32_to_bf16<<<(N * 64 + 255) / 256, 256, 0, stream>>>(x, hbf, N * 64);
    w_transpose_bf16<<<(128 * 256 + 255) / 256, 256, 0, stream>>>(W1, wtb, 128, 256);
    gemm_bf16_wmma<128><<<dim3(mblocks, 4), gblk, 0, stream>>>(hbf, wtb, bufA, N, 256);
    run_edge_phase(bufA, bufB, a1s, a1d, b1, 4, 64, /*ELU*/0,
                   al_s, al_d, mbuf, dbuf, src, dst, N, E, stream);

    // ---- Layer 2: h[N,256] @ W2[256,256] ----
    f32_to_bf16<<<(N * 128 + 255) / 256, 256, 0, stream>>>(bufB, hbf, N * 128);
    w_transpose_bf16<<<(256 * 256 + 255) / 256, 256, 0, stream>>>(W2, wtb, 256, 256);
    gemm_bf16_wmma<256><<<dim3(mblocks, 4), gblk, 0, stream>>>(hbf, wtb, bufA, N, 256);
    run_edge_phase(bufA, bufB, a2s, a2d, b2, 4, 64, /*ELU*/0,
                   al_s, al_d, mbuf, dbuf, src, dst, N, E, stream);

    // ---- Layer 3: h[N,256] @ W3[256,64] ----
    f32_to_bf16<<<(N * 128 + 255) / 256, 256, 0, stream>>>(bufB, hbf, N * 128);
    w_transpose_bf16<<<(256 * 64 + 255) / 256, 256, 0, stream>>>(W3, wtb, 256, 64);
    gemm_bf16_wmma<256><<<dim3(mblocks, 1), gblk, 0, stream>>>(hbf, wtb, bufA, N, 64);
    run_edge_phase(bufA, bufB, a3s, a3d, b3, 1, 64, /*ReLU*/1,
                   al_s, al_d, mbuf, dbuf, src, dst, N, E, stream);

    // ---- MLP head + log_softmax ----
    classifier<<<(N + 127) / 128, 128, 0, stream>>>(bufB, cW1, cb1, cW2, cb2, out, N);
}
